// CausalSelfAttention_88098369176359
// MI455X (gfx1250) — matrix operand order repaired and checked by reading, not compiled, not run on hardware
//
#include <hip/hip_runtime.h>
#include <math.h>

typedef __bf16 bf16_t;
typedef __attribute__((ext_vector_type(16))) __bf16 v16bf;
typedef __attribute__((ext_vector_type(8)))  __bf16 v8bf;
typedef __attribute__((ext_vector_type(8)))  float  v8f;
typedef __attribute__((ext_vector_type(4)))  float  v4f;

#define TT 4096
#define CC 768
#define NH 12
#define HD 64

// softmax scale folded to log2 domain: applied to Q once at production
#define QSCALE (0.125f * 1.4426950408889634f)

__device__ __forceinline__ bf16_t f2bf(float f) { return (bf16_t)f; }

__device__ __forceinline__ v16bf cat8(v8bf lo, v8bf hi) {
  return __builtin_shufflevector(lo, hi, 0,1,2,3,4,5,6,7,8,9,10,11,12,13,14,15);
}

// A fragment (16x32 bf16): lane<16 holds K[0..7],K[16..23]; lane>=16 K[8..15],K[24..31]
__device__ __forceinline__ v16bf a_frag_bf16(const bf16_t* row, bool hi_half) {
  int k0 = hi_half ? 8 : 0;
  v8bf lo = *(const v8bf*)(row + k0);
  v8bf hi = *(const v8bf*)(row + k0 + 16);
  return cat8(lo, hi);
}

// B fragment (32x16 bf16): lane<16 holds K[0..7],K[16..23]; lane>=16 K[8..15],K[24..31] (col-major source)
__device__ __forceinline__ v16bf b_frag_bf16(const bf16_t* col, bool hi_half) {
  int k0 = hi_half ? 8 : 0;
  v8bf lo = *(const v8bf*)(col + k0);
  v8bf hi = *(const v8bf*)(col + k0 + 16);
  return cat8(lo, hi);
}

__device__ __forceinline__ v8f wmma_bf16(v16bf a, v16bf b, v8f c) {
  return __builtin_amdgcn_wmma_f32_16x16x32_bf16(false, a, false, b, (short)0, c, false, false);
}

// ---------------- weight transpose + fp32->bf16 convert: src[R][Cc] -> dst[Cc][R]
__global__ void transpose_cvt_kernel(const float* __restrict__ src,
                                     bf16_t* __restrict__ dst, int R, int Cc) {
  __shared__ float tile[32][33];
  int c0 = blockIdx.x * 32, r0 = blockIdx.y * 32;
  #pragma unroll
  for (int j = 0; j < 4; j++) {
    int r = r0 + threadIdx.y + j * 8;
    tile[threadIdx.y + j * 8][threadIdx.x] = src[(size_t)r * Cc + c0 + threadIdx.x];
  }
  __syncthreads();
  #pragma unroll
  for (int j = 0; j < 4; j++) {
    int c = c0 + threadIdx.y + j * 8;
    dst[(size_t)c * R + r0 + threadIdx.x] = f2bf(tile[threadIdx.x][threadIdx.y + j * 8]);
  }
}

// ---------------- QKV GEMM: qkv = x @ w_attn + b ; scatter to q[h][t][d] (pre-scaled),
// k[h][t][d], vT[h][d][t].  Block tile 128x128, wave tile 32x64.
__global__ __launch_bounds__(256) void qkv_kernel(
    const float* __restrict__ x, const bf16_t* __restrict__ wT,
    const float* __restrict__ bias,
    bf16_t* __restrict__ qb, bf16_t* __restrict__ kb, bf16_t* __restrict__ vT) {
  int wave = threadIdx.x >> 5, lane = threadIdx.x & 31;
  int lm = lane & 15; bool hi = lane >= 16;
  int wm = wave >> 1, wn = wave & 1;
  int m0 = blockIdx.y * 128 + wm * 32;
  int n0 = blockIdx.x * 128 + wn * 64;
  v8f acc[2][4] = {};
  for (int kk = 0; kk < CC; kk += 32) {
    v16bf af[2];
    #pragma unroll
    for (int mt = 0; mt < 2; mt++) {
      const float* rp = x + (size_t)(m0 + mt * 16 + lm) * CC + kk;
      int k0 = hi ? 8 : 0;
      v4f f0 = *(const v4f*)(rp + k0);
      v4f f1 = *(const v4f*)(rp + k0 + 4);
      v4f f2 = *(const v4f*)(rp + k0 + 16);
      v4f f3 = *(const v4f*)(rp + k0 + 20);
      v16bf a;
      #pragma unroll
      for (int i = 0; i < 4; i++) {
        a[i]      = f2bf(f0[i]);
        a[4 + i]  = f2bf(f1[i]);
        a[8 + i]  = f2bf(f2[i]);
        a[12 + i] = f2bf(f3[i]);
      }
      af[mt] = a;
    }
    v16bf bf[4];
    #pragma unroll
    for (int nt = 0; nt < 4; nt++)
      bf[nt] = b_frag_bf16(wT + (size_t)(n0 + nt * 16 + lm) * CC + kk, hi);
    #pragma unroll
    for (int mt = 0; mt < 2; mt++)
      #pragma unroll
      for (int nt = 0; nt < 4; nt++)
        acc[mt][nt] = wmma_bf16(af[mt], bf[nt], acc[mt][nt]);
  }
  #pragma unroll
  for (int nt = 0; nt < 4; nt++) {
    int n = n0 + nt * 16 + lm;
    float bv = bias[n];
    int which = n / CC;
    int ci = n - which * CC;
    int h = ci >> 6, d = ci & 63;
    #pragma unroll
    for (int mt = 0; mt < 2; mt++) {
      #pragma unroll
      for (int e = 0; e < 8; e++) {
        int t = m0 + mt * 16 + e + (hi ? 8 : 0);
        float v = acc[mt][nt][e] + bv;
        if (which == 0)      qb[((size_t)h * TT + t) * HD + d] = f2bf(v * QSCALE);
        else if (which == 1) kb[((size_t)h * TT + t) * HD + d] = f2bf(v);
        else                 vT[((size_t)h * HD + d) * TT + t] = f2bf(v);
      }
    }
  }
}

// ---------------- flash attention: block = 128 query rows of one head, wave = 16 rows
__global__ __launch_bounds__(256) void attn_kernel(
    const bf16_t* __restrict__ qb, const bf16_t* __restrict__ kb,
    const bf16_t* __restrict__ vT, bf16_t* __restrict__ y) {
  __shared__ bf16_t pbuf[8][16][136];   // per-wave P tile, 272B row stride (16B aligned, conflict-free)
  int wave = threadIdx.x >> 5, lane = threadIdx.x & 31;
  int lm = lane & 15; bool hi = lane >= 16;
  int qblk = blockIdx.x, h = blockIdx.y;
  int q0 = qblk * 128 + wave * 16;
  const size_t hbase = (size_t)h * TT;

  v16bf qf[2];
  #pragma unroll
  for (int c = 0; c < 2; c++)
    qf[c] = a_frag_bf16(qb + (hbase + q0 + lm) * HD + 32 * c, hi);

  // ones B-fragment: osum = P @ 1  gives the softmax row-sum as a WMMA output
  v16bf ones;
  #pragma unroll
  for (int i = 0; i < 16; i++) ones[i] = f2bf(1.0f);

  float m_run[8];
  #pragma unroll
  for (int e = 0; e < 8; e++) m_run[e] = -1e30f;
  v8f o[4] = {};
  v8f osum = {};

  bf16_t (*pw)[136] = pbuf[wave];

  for (int kbk = 0; kbk <= qblk; kbk++) {
    int s0 = kbk * 128;
    v8f st[8];
    #pragma unroll
    for (int nt = 0; nt < 8; nt++) {
      v8f a = {};
      #pragma unroll
      for (int c = 0; c < 2; c++) {
        v16bf kf = b_frag_bf16(kb + (hbase + s0 + nt * 16 + lm) * HD + 32 * c, hi);
        a = wmma_bf16(qf[c], kf, a);
      }
      st[nt] = a;
    }
    // causal mask on the diagonal block (Q pre-scaled, S already in log2 domain)
    if (kbk == qblk) {
      #pragma unroll
      for (int nt = 0; nt < 8; nt++) {
        int scol = s0 + nt * 16 + lm;
        #pragma unroll
        for (int e = 0; e < 8; e++) {
          int trow = q0 + e + (hi ? 8 : 0);
          if (scol > trow) st[nt][e] = -1e30f;
        }
      }
    }
    float alpha[8], mnew[8];
    #pragma unroll
    for (int e = 0; e < 8; e++) {
      float mx = st[0][e];
      #pragma unroll
      for (int nt = 1; nt < 8; nt++) mx = fmaxf(mx, st[nt][e]);
      mx = fmaxf(mx, __shfl_xor(mx, 1));
      mx = fmaxf(mx, __shfl_xor(mx, 2));
      mx = fmaxf(mx, __shfl_xor(mx, 4));
      mx = fmaxf(mx, __shfl_xor(mx, 8));
      mnew[e] = fmaxf(m_run[e], mx);
      alpha[e] = exp2f(m_run[e] - mnew[e]);
      m_run[e] = mnew[e];
    }
    #pragma unroll
    for (int nt = 0; nt < 8; nt++) {
      #pragma unroll
      for (int e = 0; e < 8; e++) {
        float p = exp2f(st[nt][e] - mnew[e]);
        pw[e + (hi ? 8 : 0)][nt * 16 + lm] = f2bf(p);
      }
    }
    #pragma unroll
    for (int dt = 0; dt < 4; dt++)
      #pragma unroll
      for (int e = 0; e < 8; e++) o[dt][e] *= alpha[e];
    #pragma unroll
    for (int e = 0; e < 8; e++) osum[e] *= alpha[e];
    // P @ [V | 1]  (P via per-wave LDS reshape C-layout -> A-layout; V pre-transposed [h][d][t])
    #pragma unroll
    for (int c = 0; c < 4; c++) {
      v16bf pf = a_frag_bf16(&pw[lm][32 * c], hi);
      osum = wmma_bf16(pf, ones, osum);
      #pragma unroll
      for (int dt = 0; dt < 4; dt++) {
        v16bf vf = b_frag_bf16(vT + ((size_t)h * HD + dt * 16 + lm) * TT + s0 + 32 * c, hi);
        o[dt] = wmma_bf16(pf, vf, o[dt]);
      }
    }
  }
  #pragma unroll
  for (int dt = 0; dt < 4; dt++) {
    #pragma unroll
    for (int e = 0; e < 8; e++) {
      int t = q0 + e + (hi ? 8 : 0);
      int col = h * HD + dt * 16 + lm;
      y[(size_t)t * CC + col] = f2bf(o[dt][e] / osum[e]);
    }
  }
}

// ---------------- output projection: out = y @ w_proj + b (fp32 out)
// block tile 128x128, wave tile 32x64
__global__ __launch_bounds__(256) void proj_kernel(
    const bf16_t* __restrict__ y, const bf16_t* __restrict__ wT,
    const float* __restrict__ bias, float* __restrict__ out) {
  int wave = threadIdx.x >> 5, lane = threadIdx.x & 31;
  int lm = lane & 15; bool hi = lane >= 16;
  int wm = wave >> 1, wn = wave & 1;
  int m0 = blockIdx.y * 128 + wm * 32;
  int n0 = blockIdx.x * 128 + wn * 64;
  v8f acc[2][4] = {};
  for (int kk = 0; kk < CC; kk += 32) {
    v16bf af[2], bf[4];
    #pragma unroll
    for (int mt = 0; mt < 2; mt++)
      af[mt] = a_frag_bf16(y + (size_t)(m0 + mt * 16 + lm) * CC + kk, hi);
    #pragma unroll
    for (int nt = 0; nt < 4; nt++)
      bf[nt] = b_frag_bf16(wT + (size_t)(n0 + nt * 16 + lm) * CC + kk, hi);
    #pragma unroll
    for (int mt = 0; mt < 2; mt++)
      #pragma unroll
      for (int nt = 0; nt < 4; nt++)
        acc[mt][nt] = wmma_bf16(af[mt], bf[nt], acc[mt][nt]);
  }
  #pragma unroll
  for (int nt = 0; nt < 4; nt++) {
    int n = n0 + nt * 16 + lm;
    float bv = bias[n];
    #pragma unroll
    for (int mt = 0; mt < 2; mt++)
      #pragma unroll
      for (int e = 0; e < 8; e++) {
        int t = m0 + mt * 16 + e + (hi ? 8 : 0);
        out[(size_t)t * CC + n] = acc[mt][nt][e] + bv;
      }
  }
}

extern "C" void kernel_launch(void* const* d_in, const int* in_sizes, int n_in,
                              void* d_out, int out_size, void* d_ws, size_t ws_size,
                              hipStream_t stream) {
  const float* x      = (const float*)d_in[0];
  const float* w_attn = (const float*)d_in[1];
  const float* b_attn = (const float*)d_in[2];
  const float* w_proj = (const float*)d_in[3];
  const float* b_proj = (const float*)d_in[4];
  float* out = (float*)d_out;

  bf16_t* wattnT = (bf16_t*)d_ws;                    // [3C][C]
  bf16_t* wprojT = wattnT + (size_t)CC * 3 * CC;     // [C][C]
  bf16_t* qb     = wprojT + (size_t)CC * CC;         // [H][T][D] (pre-scaled)
  bf16_t* kb     = qb + (size_t)TT * CC;             // [H][T][D]
  bf16_t* vT     = kb + (size_t)TT * CC;             // [H][D][T]
  bf16_t* ybuf   = vT + (size_t)TT * CC;             // [T][C]

  transpose_cvt_kernel<<<dim3(3 * CC / 32, CC / 32), dim3(32, 8), 0, stream>>>(w_attn, wattnT, CC, 3 * CC);
  transpose_cvt_kernel<<<dim3(CC / 32, CC / 32), dim3(32, 8), 0, stream>>>(w_proj, wprojT, CC, CC);
  qkv_kernel<<<dim3(3 * CC / 128, TT / 128), 256, 0, stream>>>(x, wattnT, b_attn, qb, kb, vT);
  attn_kernel<<<dim3(TT / 128, NH), 256, 0, stream>>>(qb, kb, vT, ybuf);
  proj_kernel<<<dim3(CC / 128, TT / 128), 256, 0, stream>>>(ybuf, wprojT, b_proj, out);
}
